// SearchTransfer_41128606827034
// MI455X (gfx1250) — compile-verified, matrix operand order checked by reading, not run on hardware
//
#include <hip/hip_runtime.h>
#include <hip/hip_bf16.h>

// ---------------------------------------------------------------------------
// SearchTransfer (TTSR) for MI455X / gfx1250, wave32 + WMMA f16 + async-LDS.
//   inputs : lrsr_lv3 (1,256,48,48), refsr_lv3 (1,256,48,48),
//            ref_lv1 (1,64,192,192), ref_lv2 (1,128,96,96), ref_lv3 (1,256,48,48)
//   outputs: S (1,1,48,48) | T_lv3 (1,256,48,48) | T_lv2 (1,128,96,96) | T_lv1 (1,64,192,192)
// ---------------------------------------------------------------------------

typedef __attribute__((ext_vector_type(16))) _Float16 v16h;
typedef __attribute__((ext_vector_type(8)))  _Float16 v8h;
typedef __attribute__((ext_vector_type(8)))  float    v8f;
typedef __attribute__((ext_vector_type(4)))  int      v4i;

#define KD        2304   // feature length = 256*3*3
#define NPATCH    2304   // 48*48 patches
#define GRID48    48
#define ROWSPLIT  4      // row-dimension split of the argmax-GEMM
#define ROWS_PER  (NPATCH / ROWSPLIT)   // 576

#define AS_GLOBAL __attribute__((address_space(1)))
#define AS_LDS    __attribute__((address_space(3)))

#if defined(__has_builtin)
#if __has_builtin(__builtin_amdgcn_global_load_async_to_lds_b128)
#define ASYNC_LDS 1
#endif
#endif

// ---------------------------------------------------------------------------
// Kernel 1: unfold(3x3, pad 1) + L2-normalize each patch vector, emit f16
// row-major [patch][feature]. One block per patch; 256 threads x 9 features.
// ---------------------------------------------------------------------------
__global__ __launch_bounds__(256) void pack_normalize_k(
    const float* __restrict__ x,      // (256,48,48)
    _Float16* __restrict__ P)         // [2304][2304] f16
{
  const int patch = blockIdx.x;
  const int oy = patch / GRID48, ox = patch % GRID48;

  float v[9];
  float ss = 0.f;
#pragma unroll
  for (int j = 0; j < 9; ++j) {
    const int f = threadIdx.x + j * 256;          // 9*256 == 2304 exactly
    const int c = f / 9, rem = f % 9;
    const int y  = oy + rem / 3 - 1;
    const int xx = ox + rem % 3 - 1;
    float val = 0.f;
    if ((unsigned)y < 48u && (unsigned)xx < 48u)
      val = x[(c * GRID48 + y) * GRID48 + xx];
    v[j] = val;
    ss += val * val;
  }

  // wave32 reduce, then cross-wave reduce in LDS
  for (int off = 16; off > 0; off >>= 1) ss += __shfl_down(ss, off);
  __shared__ float wsum[8];
  if ((threadIdx.x & 31) == 0) wsum[threadIdx.x >> 5] = ss;
  __syncthreads();
  if (threadIdx.x == 0) {
    float t = 0.f;
#pragma unroll
    for (int i = 0; i < 8; ++i) t += wsum[i];
    wsum[0] = 1.0f / fmaxf(sqrtf(t), 1e-12f);
  }
  __syncthreads();
  const float scale = wsum[0];

#pragma unroll
  for (int j = 0; j < 9; ++j) {
    const int f = threadIdx.x + j * 256;
    P[(size_t)patch * KD + f] = (_Float16)(v[j] * scale);
  }
}

// ---------------------------------------------------------------------------
// Fragment load: 16 f16 per lane as two 16-byte chunks (ISA 16-bit 16x32
// layout: VGPR0-3 = K(half)+0..7, VGPR4-7 = K(half)+16..23). `base` already
// includes the lane's K-half offset (0 or 8).
// ---------------------------------------------------------------------------
__device__ __forceinline__ v16h load_frag16(const _Float16* base, int k0)
{
  union { v16h v; v8h h[2]; } u;
  u.h[0] = *(const v8h*)(base + k0);
  u.h[1] = *(const v8h*)(base + k0 + 16);
  return u.v;
}

// ---------------------------------------------------------------------------
// Kernel 2: fused GEMM (A = normalized ref patches [row][k],
// Bt = normalized lr patches [col][k]) + per-column max/argmax over a row
// slice. Grid = (72 col-blocks, 4 row-splits). Block: 8 waves, 32 columns,
// B panel (144 KB) staged in LDS via GLOBAL_LOAD_ASYNC_TO_LDS_B128
// (ASYNCcnt protocol: s_wait_asynccnt 0 + barrier). Each wave:
// colTile = wave/4, rowGroup = wave%4; 9 passes x 72 K-steps of
// v_wmma_f32_16x16x32_f16 per row-split.
// ---------------------------------------------------------------------------
__global__ __launch_bounds__(256) void gemm_argmax_k(
    const _Float16* __restrict__ A,
    const _Float16* __restrict__ Bt,
    float* __restrict__ pval,        // [ROWSPLIT][NPATCH] partial max
    int*   __restrict__ pidx)        // [ROWSPLIT][NPATCH] partial argmax
{
  __shared__ __align__(16) _Float16 Bs[32 * KD];
  __shared__ float sv[8][16];
  __shared__ int   si[8][16];

  const int colBase  = blockIdx.x * 32;
  const int rowSplit = blockIdx.y;
  const int rowOrg   = rowSplit * ROWS_PER;

  // cooperative B-panel stage: 32 cols * 288 16-byte chunks
  for (int i = threadIdx.x; i < 32 * (KD / 8); i += 256) {
    const int col = i / (KD / 8);
    const int kk  = (i % (KD / 8)) * 8;
    const _Float16* gsrc = &Bt[(size_t)(colBase + col) * KD + kk];
    _Float16*       ldst = &Bs[col * KD + kk];
#if ASYNC_LDS
    __builtin_amdgcn_global_load_async_to_lds_b128(
        (AS_GLOBAL v4i*)gsrc, (AS_LDS v4i*)ldst, 0, 0);
#else
    *(v8h*)ldst = *(const v8h*)gsrc;
#endif
  }
#if ASYNC_LDS
#if defined(__has_builtin) && __has_builtin(__builtin_amdgcn_s_wait_asynccnt)
  __builtin_amdgcn_s_wait_asynccnt(0);
#else
  asm volatile("s_wait_asynccnt 0x0" ::: "memory");
#endif
#endif
  __syncthreads();

  const int lane     = threadIdx.x & 31;
  const int wave     = threadIdx.x >> 5;
  const int colTile  = wave >> 2;          // 0..1
  const int rowGroup = wave & 3;           // 0..3
  const int lane16   = lane & 15;
  const int khalf    = (lane < 16) ? 0 : 8;
  const int mhalf    = (lane < 16) ? 0 : 8;

  const _Float16* bbase = &Bs[(colTile * 16 + lane16) * KD + khalf];

  float bestV = -3.0e38f;
  int   bestI = 0;

  for (int rowBase = 0; rowBase < ROWS_PER; rowBase += 64) {
    const int aRow = rowOrg + rowBase + rowGroup * 16 + lane16;  // A: M = lane%16
    const _Float16* abase = &A[(size_t)aRow * KD + khalf];

    v8f acc = {};
#pragma unroll 4
    for (int k0 = 0; k0 < KD; k0 += 32) {
      v16h a = load_frag16(abase, k0);
      v16h b = load_frag16(bbase, k0);
      acc = __builtin_amdgcn_wmma_f32_16x16x32_f16(
          /*neg_a=*/false, a, /*neg_b=*/false, b,
          /*c_mod=*/(short)0, acc, /*reuse_a=*/false, /*reuse_b=*/false);
    }

    // C layout: lane column n = lane%16; VGPR e -> row mhalf+e
#pragma unroll
    for (int e = 0; e < 8; ++e) {
      const float cv = acc[e];
      const int r = rowOrg + rowBase + rowGroup * 16 + mhalf + e;
      if (cv > bestV) { bestV = cv; bestI = r; }   // strict > keeps lowest row
    }
  }

  // lanes L and L+16 hold the same column -> xor-16 reduce (tie: lower index)
  {
    const float ov = __shfl_xor(bestV, 16);
    const int   oi = __shfl_xor(bestI, 16);
    if (ov > bestV || (ov == bestV && oi < bestI)) { bestV = ov; bestI = oi; }
  }

  if (lane < 16) { sv[wave][lane] = bestV; si[wave][lane] = bestI; }
  __syncthreads();

  if (threadIdx.x < 32) {
    const int ct = threadIdx.x >> 4;   // col tile
    const int n  = threadIdx.x & 15;
    float bv = sv[ct * 4 + 0][n];
    int   bi = si[ct * 4 + 0][n];
#pragma unroll
    for (int w = 1; w < 4; ++w) {
      const float v2 = sv[ct * 4 + w][n];
      const int   i2 = si[ct * 4 + w][n];
      if (v2 > bv || (v2 == bv && i2 < bi)) { bv = v2; bi = i2; }
    }
    const int col = colBase + ct * 16 + n;
    pval[rowSplit * NPATCH + col] = bv;
    pidx[rowSplit * NPATCH + col] = bi;
  }
}

// ---------------------------------------------------------------------------
// Kernel 2b: reduce the ROWSPLIT partial argmax results per column.
// ---------------------------------------------------------------------------
__global__ __launch_bounds__(256) void final_argmax_k(
    const float* __restrict__ pval, const int* __restrict__ pidx,
    float* __restrict__ Rstar, int* __restrict__ Rarg)
{
  const int col = blockIdx.x * blockDim.x + threadIdx.x;
  if (col >= NPATCH) return;
  float bv = pval[col];
  int   bi = pidx[col];
#pragma unroll
  for (int s = 1; s < ROWSPLIT; ++s) {
    const float v2 = pval[s * NPATCH + col];
    const int   i2 = pidx[s * NPATCH + col];
    if (v2 > bv || (v2 == bv && i2 < bi)) { bv = v2; bi = i2; }
  }
  Rstar[col] = bv;
  Rarg[col]  = bi;
}

// ---------------------------------------------------------------------------
// Kernel 3: fused take_along_axis + fold + /9. For each output pixel (c,y,x),
// sum the <=9 valid (ki,kj) taps: patch (oy,ox) with oy*stride+ki-pad == y,
// gathering ref[c, py*stride+ki-pad, px*stride+kj-pad] at p = arg[oy*48+ox].
// ---------------------------------------------------------------------------
__global__ __launch_bounds__(256) void fold_gather_k(
    const float* __restrict__ ref,    // (C,H,H)
    const int*   __restrict__ arg,    // 2304 patch indices
    float* __restrict__ out,          // (C,H,H)
    int C, int H, int k, int pad, int stride)
{
  const int i = blockIdx.x * blockDim.x + threadIdx.x;
  if (i >= C * H * H) return;
  const int x = i % H;
  const int y = (i / H) % H;
  const int c = i / (H * H);

  float acc = 0.f;
  for (int ki = 0; ki < k; ++ki) {
    const int ty = y + pad - ki;
    if (ty < 0 || (ty % stride) != 0) continue;
    const int oy = ty / stride;
    if (oy >= GRID48) continue;
    for (int kj = 0; kj < k; ++kj) {
      const int tx = x + pad - kj;
      if (tx < 0 || (tx % stride) != 0) continue;
      const int ox = tx / stride;
      if (ox >= GRID48) continue;
      const int p  = arg[oy * GRID48 + ox];
      const int py = p / GRID48, px = p % GRID48;
      const int sy = py * stride + ki - pad;
      const int sx = px * stride + kj - pad;
      if ((unsigned)sy < (unsigned)H && (unsigned)sx < (unsigned)H)
        acc += ref[(c * H + sy) * H + sx];
    }
  }
  out[i] = acc * (1.0f / 9.0f);
}

// ---------------------------------------------------------------------------
extern "C" void kernel_launch(void* const* d_in, const int* in_sizes, int n_in,
                              void* d_out, int out_size, void* d_ws, size_t ws_size,
                              hipStream_t stream)
{
  const float* lrsr  = (const float*)d_in[0];   // (1,256,48,48)
  const float* refsr = (const float*)d_in[1];   // (1,256,48,48)
  const float* ref1  = (const float*)d_in[2];   // (1,64,192,192)
  const float* ref2  = (const float*)d_in[3];   // (1,128,96,96)
  const float* ref3  = (const float*)d_in[4];   // (1,256,48,48)

  float* S  = (float*)d_out;                    // 2304
  float* T3 = S + 2304;                         // 256*48*48   = 589824
  float* T2 = T3 + 256 * 48 * 48;               // 128*96*96   = 1179648
  float* T1 = T2 + 128 * 96 * 96;               // 64*192*192  = 2359296

  _Float16* A    = (_Float16*)d_ws;                         // ref patches, 10.6 MB
  _Float16* Bm   = A + (size_t)NPATCH * KD;                 // lr  patches, 10.6 MB
  int*      Rarg = (int*)(Bm + (size_t)NPATCH * KD);        // 9 KB
  float*    pval = (float*)(Rarg + NPATCH);                 // 36 KB
  int*      pidx = (int*)(pval + ROWSPLIT * NPATCH);        // 36 KB

  // 1) build normalized patch matrices (A = ref rows, Bm = lr cols)
  pack_normalize_k<<<NPATCH, 256, 0, stream>>>(refsr, A);
  pack_normalize_k<<<NPATCH, 256, 0, stream>>>(lrsr,  Bm);

  // 2) fused WMMA correlation + per-column max/argmax (row-split x4)
  gemm_argmax_k<<<dim3(NPATCH / 32, ROWSPLIT), 256, 0, stream>>>(A, Bm, pval, pidx);
  final_argmax_k<<<(NPATCH + 255) / 256, 256, 0, stream>>>(pval, pidx, S, Rarg);

  // 3) fused gather + fold at the three pyramid levels
  {
    const int n3 = 256 * 48 * 48;
    fold_gather_k<<<(n3 + 255) / 256, 256, 0, stream>>>(ref3, Rarg, T3, 256, 48, 3, 1, 1);
    const int n2 = 128 * 96 * 96;
    fold_gather_k<<<(n2 + 255) / 256, 256, 0, stream>>>(ref2, Rarg, T2, 128, 96, 6, 2, 2);
    const int n1 = 64 * 192 * 192;
    fold_gather_k<<<(n1 + 255) / 256, 256, 0, stream>>>(ref1, Rarg, T1, 64, 192, 12, 4, 4);
  }
}
